// MultiHeadAttention_59803124629926
// MI455X (gfx1250) — matrix operand order repaired and checked by reading, not compiled, not run on hardware
//
#include <hip/hip_runtime.h>
#include <math.h>

#define DMODEL 1024
#define SEQ    1024
#define NHEADS 16
#define HDEPTH 64
#define NBATCH 4

typedef __attribute__((ext_vector_type(16))) __bf16 v16bf;
typedef __attribute__((ext_vector_type(8)))  __bf16 v8bf;
typedef __attribute__((ext_vector_type(4)))  __bf16 v4bf;
typedef __attribute__((ext_vector_type(8)))  float  v8f;

__device__ __forceinline__ v16bf mk16(v8bf lo, v8bf hi) {
  v16bf r;
#pragma unroll
  for (int i = 0; i < 8; ++i) { r[i] = lo[i]; r[i + 8] = hi[i]; }
  return r;
}
__device__ __forceinline__ v8bf ldb8(const __bf16* p) { return *(const v8bf*)p; }

// ---------------------------------------------------------------- fp32 -> bf16
__global__ __launch_bounds__(256) void k_cvt_bf16(const float* __restrict__ in,
                                                  __bf16* __restrict__ out) {
  int i = (blockIdx.x * 256 + threadIdx.x) * 4;
  float4 v = *(const float4*)(in + i);
  v4bf o; o[0] = (__bf16)v.x; o[1] = (__bf16)v.y; o[2] = (__bf16)v.z; o[3] = (__bf16)v.w;
  *(v4bf*)(out + i) = o;
}

// -------------------------------------------- weight transpose + cvt: wt[n][k]=w[k][n]
__global__ __launch_bounds__(256) void k_wtrans(const float* __restrict__ w,
                                                __bf16* __restrict__ wt) {
  __shared__ float tile[32][33];
  const int bx = blockIdx.x * 32, by = blockIdx.y * 32;
  const int tx = threadIdx.x, ty = threadIdx.y;
#pragma unroll
  for (int j = ty; j < 32; j += 8) tile[j][tx] = w[(by + j) * DMODEL + bx + tx];
  __syncthreads();
#pragma unroll
  for (int j = ty; j < 32; j += 8)
    wt[(long)(bx + j) * DMODEL + by + tx] = (__bf16)tile[tx][j];
}

// ---- shared fragment-load + WMMA helpers for the 16x64 per-wave GEMM strip ----
__device__ __forceinline__ void load_frags(const __bf16* xrow, const __bf16* wb,
                                           int kk, v16bf& a, v16bf b[4]) {
  a = mk16(ldb8(xrow + kk), ldb8(xrow + kk + 16));
#pragma unroll
  for (int t = 0; t < 4; ++t) {
    const __bf16* p = wb + (long)t * 16 * DMODEL + kk;
    b[t] = mk16(ldb8(p), ldb8(p + 16));
  }
}
__device__ __forceinline__ void do_wmma4(const v16bf& a, const v16bf b[4], v8f acc[4]) {
#pragma unroll
  for (int t = 0; t < 4; ++t)
    acc[t] = __builtin_amdgcn_wmma_f32_16x16x32_bf16(false, a, false, b[t],
                                                     (short)0, acc[t], false, false);
}

// ------------------------------------------------- bf16 WMMA GEMM: projections
// Ping-pong double buffering (2x unrolled): no loop-carried register copies;
// each half-step issues the next fragment loads, then runs 4 WMMAs on the
// other buffer, so the pre-WMMA wait only covers loads already in flight.
// mode 0: out[((b*16+h)*1024+s)*64+dd]  (Q,K layout (B,H,S,64))
// mode 1: out[((b*16+h)*64+dd)*1024+s]  (V transposed  (B,H,64,S))
__global__ __launch_bounds__(256) void k_proj(const __bf16* __restrict__ X,
                                              const __bf16* __restrict__ Wt,
                                              const float* __restrict__ bias,
                                              __bf16* __restrict__ out, int mode) {
  const int lane = threadIdx.x & 31;
  const int wave = threadIdx.x >> 5;
  const int m0 = blockIdx.x * 128 + wave * 16;
  const int n0 = blockIdx.y * 64;
  const int l15 = lane & 15, hi = lane >> 4;
  const int koffA = hi * 8, koffB = hi * 8;
  const __bf16* xrow = X + (long)(m0 + l15) * DMODEL + koffA;
  const __bf16* wb   = Wt + (long)(n0 + l15) * DMODEL + koffB;
  v8f acc[4];
#pragma unroll
  for (int t = 0; t < 4; ++t) acc[t] = (v8f){};
  v16bf a0, b0[4], a1, b1[4];
  load_frags(xrow, wb, 0, a0, b0);
  for (int kk = 0; kk < DMODEL - 64; kk += 64) {
    __builtin_prefetch((const void*)(xrow + kk + 256), 0, 1);  // A stream ahead
    load_frags(xrow, wb, kk + 32, a1, b1);   // fill pong
    do_wmma4(a0, b0, acc);                   // compute ping
    load_frags(xrow, wb, kk + 64, a0, b0);   // refill ping
    do_wmma4(a1, b1, acc);                   // compute pong
  }
  load_frags(xrow, wb, DMODEL - 32, a1, b1); // last step
  do_wmma4(a0, b0, acc);
  do_wmma4(a1, b1, acc);
#pragma unroll
  for (int t = 0; t < 4; ++t) {
    const int n = n0 + 16 * t + l15;
    const float bv = bias[n];
    const int h = n >> 6, dd = n & 63;
#pragma unroll
    for (int r = 0; r < 8; ++r) {
      const int m = m0 + r + (hi << 3);
      const int b = m >> 10, s = m & 1023;
      const float val = acc[t][r] + bv;
      long idx = (mode == 0) ? ((long)((b * NHEADS + h) * SEQ + s) * HDEPTH + dd)
                             : ((long)((b * NHEADS + h) * HDEPTH + dd) * SEQ + s);
      out[idx] = (__bf16)val;
    }
  }
}

// ---------------------- logits = gelu(Q K^T / 8) + causal mask (lower tiles only)
__global__ __launch_bounds__(256) void k_logits(const __bf16* __restrict__ Q,
                                                const __bf16* __restrict__ K,
                                                float* __restrict__ L) {
  const int bh = blockIdx.z;
  const int wave = threadIdx.x >> 5;
  const int lane = threadIdx.x & 31;
  const int m0 = blockIdx.x * 128 + wave * 16;
  const int n0 = blockIdx.y * 64;
  if (n0 > m0 + 15) return;  // wave-uniform: strip entirely above diagonal
  const int l15 = lane & 15, hi = lane >> 4;
  const int koffA = hi * 8, koffB = hi * 8;
  const __bf16* qrow = Q + ((long)bh * SEQ + m0 + l15) * HDEPTH + koffA;
  const __bf16* kb   = K + ((long)bh * SEQ + n0 + l15) * HDEPTH + koffB;
  // preload everything (K-depth is only 64 = 2 steps), then 8 WMMAs
  v16bf a0 = mk16(ldb8(qrow), ldb8(qrow + 16));
  v16bf a1 = mk16(ldb8(qrow + 32), ldb8(qrow + 48));
  v8f acc[4];
#pragma unroll
  for (int t = 0; t < 4; ++t) acc[t] = (v8f){};
  v16bf b0[4], b1[4];
#pragma unroll
  for (int t = 0; t < 4; ++t) {
    const __bf16* p = kb + (long)t * 16 * HDEPTH;
    b0[t] = mk16(ldb8(p), ldb8(p + 16));
    b1[t] = mk16(ldb8(p + 32), ldb8(p + 48));
  }
#pragma unroll
  for (int t = 0; t < 4; ++t)
    acc[t] = __builtin_amdgcn_wmma_f32_16x16x32_bf16(false, a0, false, b0[t],
                                                     (short)0, acc[t], false, false);
#pragma unroll
  for (int t = 0; t < 4; ++t)
    acc[t] = __builtin_amdgcn_wmma_f32_16x16x32_bf16(false, a1, false, b1[t],
                                                     (short)0, acc[t], false, false);
  float* Lb = L + (long)bh * SEQ * SEQ;
#pragma unroll
  for (int t = 0; t < 4; ++t) {
    const int n = n0 + 16 * t + l15;
#pragma unroll
    for (int r = 0; r < 8; ++r) {
      const int m = m0 + r + (hi << 3);
      float x = acc[t][r] * 0.125f;                     // 1/sqrt(64)
      float g = 0.5f * x * (1.0f + erff(x * 0.70710678118654752f));  // exact GELU
      if (n > m) g += -1e9f;                            // mask * NEG_INF
      Lb[(long)m * SEQ + n] = g;
    }
  }
}

// ---------------------------------- per-row online softmax stats: (max, 1/sum)
__global__ __launch_bounds__(256) void k_stats(const float* __restrict__ L,
                                               float2* __restrict__ stats) {
  __shared__ float smx[256], ssm[256];
  const int rid = blockIdx.x;       // bh*1024 + m
  const int m = rid & (SEQ - 1);
  const float* row = L + (long)rid * SEQ;
  const int tid = threadIdx.x;
  float mx = -3.0e38f, sm = 0.f;
  for (int n = tid; n <= m; n += 256) {
    float x = row[n];
    if (x > mx) { sm = sm * expf(mx - x) + 1.f; mx = x; }
    else        { sm += expf(x - mx); }
  }
  smx[tid] = mx; ssm[tid] = sm;
  __syncthreads();
  for (int off = 128; off; off >>= 1) {
    if (tid < off) {
      float m2 = smx[tid + off], s2 = ssm[tid + off];
      float M = fmaxf(mx, m2);
      sm = sm * expf(mx - M) + s2 * expf(m2 - M);
      mx = M; smx[tid] = mx; ssm[tid] = sm;
    }
    __syncthreads();
  }
  if (tid == 0) stats[rid] = make_float2(mx, 1.f / sm);
}

// --- normalize logits in place (= final attn, fp32) and compute O = attn @ V (WMMA)
__global__ __launch_bounds__(256) void k_attn_v(float* __restrict__ L,
                                                const float2* __restrict__ stats,
                                                const __bf16* __restrict__ Vt,
                                                __bf16* __restrict__ O) {
  const int bh = blockIdx.z;
  const int wave = threadIdx.x >> 5;
  const int lane = threadIdx.x & 31;
  const int m0 = blockIdx.x * 128 + wave * 16;
  const int l15 = lane & 15, hi = lane >> 4;
  const int koffA = hi * 8, koffB = hi * 8;
  const int row = m0 + l15;                          // this lane's attn row
  const float2 st = stats[bh * SEQ + row];
  const float rmax = st.x, rinv = st.y;
  float* Lrow = L + ((long)bh * SEQ + row) * SEQ;
  const __bf16* Vb = Vt + (long)bh * HDEPTH * SEQ + koffB;
  v8f acc[4];
#pragma unroll
  for (int t = 0; t < 4; ++t) acc[t] = (v8f){};
  const int ncomp = ((m0 + 15) / 32 + 1) * 32;       // chunks touching the triangle
  for (int nc = 0; nc < ncomp; nc += 32) {
    // issue B (V) fragment loads first: they overlap the expf chain below
    v16bf bfr[4];
#pragma unroll
    for (int t = 0; t < 4; ++t) {
      const __bf16* vrow = Vb + (long)(16 * t + l15) * SEQ + nc;
      bfr[t] = mk16(ldb8(vrow), ldb8(vrow + 16));
    }
    const int na = nc + koffA, nb = nc + 16 + koffA;
    const float4 a0 = *(const float4*)(Lrow + na);
    const float4 a1 = *(const float4*)(Lrow + na + 4);
    const float4 b0 = *(const float4*)(Lrow + nb);
    const float4 b1 = *(const float4*)(Lrow + nb + 4);
    const float xs[16] = {a0.x, a0.y, a0.z, a0.w, a1.x, a1.y, a1.z, a1.w,
                          b0.x, b0.y, b0.z, b0.w, b1.x, b1.y, b1.z, b1.w};
    float pv[16];
#pragma unroll
    for (int i = 0; i < 16; ++i) {
      const int n = (i < 8) ? (na + i) : (nb + i - 8);
      pv[i] = (n <= row) ? expf(xs[i] - rmax) * rinv : 0.f;   // softmax, masked->0
    }
    *(float4*)(Lrow + na)     = make_float4(pv[0], pv[1], pv[2], pv[3]);
    *(float4*)(Lrow + na + 4) = make_float4(pv[4], pv[5], pv[6], pv[7]);
    *(float4*)(Lrow + nb)     = make_float4(pv[8], pv[9], pv[10], pv[11]);
    *(float4*)(Lrow + nb + 4) = make_float4(pv[12], pv[13], pv[14], pv[15]);
    v16bf a;
#pragma unroll
    for (int i = 0; i < 16; ++i) a[i] = (__bf16)pv[i];
#pragma unroll
    for (int t = 0; t < 4; ++t)
      acc[t] = __builtin_amdgcn_wmma_f32_16x16x32_bf16(false, a, false, bfr[t],
                                                       (short)0, acc[t], false, false);
  }
  // zero-fill the fully masked remainder of each row (attn must be exactly 0 there)
  const float4 z = make_float4(0.f, 0.f, 0.f, 0.f);
  for (int nc = ncomp; nc < SEQ; nc += 32) {
    *(float4*)(Lrow + nc + koffA)          = z;
    *(float4*)(Lrow + nc + koffA + 4)      = z;
    *(float4*)(Lrow + nc + 16 + koffA)     = z;
    *(float4*)(Lrow + nc + 16 + koffA + 4) = z;
  }
  // store O merged (B, S, D_MODEL) as bf16 for the final projection
  const int b = bh >> 4, h = bh & 15;
#pragma unroll
  for (int t = 0; t < 4; ++t) {
    const int dd = 16 * t + l15;
#pragma unroll
    for (int r = 0; r < 8; ++r) {
      const int s = m0 + r + (hi << 3);
      O[(long)(b * SEQ + s) * DMODEL + h * HDEPTH + dd] = (__bf16)acc[t][r];
    }
  }
}

// ------------------------------------------- output projection: fp32 result
__global__ __launch_bounds__(256) void k_out_gemm(const __bf16* __restrict__ X,
                                                  const __bf16* __restrict__ Wt,
                                                  const float* __restrict__ bias,
                                                  float* __restrict__ out) {
  const int lane = threadIdx.x & 31;
  const int wave = threadIdx.x >> 5;
  const int m0 = blockIdx.x * 128 + wave * 16;
  const int n0 = blockIdx.y * 64;
  const int l15 = lane & 15, hi = lane >> 4;
  const int koffA = hi * 8, koffB = hi * 8;
  const __bf16* xrow = X + (long)(m0 + l15) * DMODEL + koffA;
  const __bf16* wb   = Wt + (long)(n0 + l15) * DMODEL + koffB;
  v8f acc[4];
#pragma unroll
  for (int t = 0; t < 4; ++t) acc[t] = (v8f){};
  v16bf a0, b0[4], a1, b1[4];
  load_frags(xrow, wb, 0, a0, b0);
  for (int kk = 0; kk < DMODEL - 64; kk += 64) {
    __builtin_prefetch((const void*)(xrow + kk + 256), 0, 1);
    load_frags(xrow, wb, kk + 32, a1, b1);
    do_wmma4(a0, b0, acc);
    load_frags(xrow, wb, kk + 64, a0, b0);
    do_wmma4(a1, b1, acc);
  }
  load_frags(xrow, wb, DMODEL - 32, a1, b1);
  do_wmma4(a0, b0, acc);
  do_wmma4(a1, b1, acc);
#pragma unroll
  for (int t = 0; t < 4; ++t) {
    const int n = n0 + 16 * t + l15;
    const float bv = bias[n];
#pragma unroll
    for (int r = 0; r < 8; ++r) {
      const int m = m0 + r + (hi << 3);
      out[(long)m * DMODEL + n] = acc[t][r] + bv;
    }
  }
}

extern "C" void kernel_launch(void* const* d_in, const int* in_sizes, int n_in,
                              void* d_out, int out_size, void* d_ws, size_t ws_size,
                              hipStream_t stream) {
  const float* x_q = (const float*)d_in[0];
  const float* x_k = (const float*)d_in[1];
  const float* x_v = (const float*)d_in[2];
  // d_in[3] = mask (unused: causality applied analytically)
  const float* wq = (const float*)d_in[4];
  const float* bq = (const float*)d_in[5];
  const float* wk = (const float*)d_in[6];
  const float* bk = (const float*)d_in[7];
  const float* wv = (const float*)d_in[8];
  const float* bv = (const float*)d_in[9];
  const float* wo = (const float*)d_in[10];
  const float* bo = (const float*)d_in[11];

  const size_t MB = 1024ull * 1024ull;
  char* w = (char*)d_ws;
  __bf16* xq_bf = (__bf16*)(w + 0 * MB);   // 8MB each
  __bf16* xk_bf = (__bf16*)(w + 8 * MB);
  __bf16* xv_bf = (__bf16*)(w + 16 * MB);
  __bf16* wqt   = (__bf16*)(w + 24 * MB);  // 2MB each
  __bf16* wkt   = (__bf16*)(w + 26 * MB);
  __bf16* wvt   = (__bf16*)(w + 28 * MB);
  __bf16* wot   = (__bf16*)(w + 30 * MB);
  __bf16* Qbf   = (__bf16*)(w + 32 * MB);  // 8MB
  __bf16* Kbf   = (__bf16*)(w + 40 * MB);  // 8MB
  __bf16* Vtb   = (__bf16*)(w + 48 * MB);  // 8MB (B,H,64,S)
  __bf16* Obf   = (__bf16*)(w + 56 * MB);  // 8MB merged (B,S,1024)
  float2* stats = (float2*)(w + 64 * MB);  // 512KB

  float* out0  = (float*)d_out;                    // (B,S,1024)
  float* attnL = (float*)d_out + 4 * SEQ * DMODEL; // (B,H,S,S): logits then attn

  const int NELM = NBATCH * SEQ * DMODEL;          // 4M
  // 1) input conversions
  k_cvt_bf16<<<NELM / 4 / 256, 256, 0, stream>>>(x_q, xq_bf);
  k_cvt_bf16<<<NELM / 4 / 256, 256, 0, stream>>>(x_k, xk_bf);
  k_cvt_bf16<<<NELM / 4 / 256, 256, 0, stream>>>(x_v, xv_bf);
  // 2) weight transposes
  dim3 tb(32, 8), tg(32, 32);
  k_wtrans<<<tg, tb, 0, stream>>>(wq, wqt);
  k_wtrans<<<tg, tb, 0, stream>>>(wk, wkt);
  k_wtrans<<<tg, tb, 0, stream>>>(wv, wvt);
  k_wtrans<<<tg, tb, 0, stream>>>(wo, wot);
  // 3) projections (M=4096 rows / 128, N=1024 / 64)
  dim3 pg(32, 16);
  k_proj<<<pg, 256, 0, stream>>>(xq_bf, wqt, bq, Qbf, 0);
  k_proj<<<pg, 256, 0, stream>>>(xk_bf, wkt, bk, Kbf, 0);
  k_proj<<<pg, 256, 0, stream>>>(xv_bf, wvt, bv, Vtb, 1);
  // 4) masked GELU logits
  dim3 lg(SEQ / 128, SEQ / 64, NBATCH * NHEADS);
  k_logits<<<lg, 256, 0, stream>>>(Qbf, Kbf, attnL);
  // 5) softmax row stats
  k_stats<<<NBATCH * NHEADS * SEQ, 256, 0, stream>>>(attnL, stats);
  // 6) normalize in place + attn @ V
  dim3 ag(SEQ / 128, 1, NBATCH * NHEADS);
  k_attn_v<<<ag, 256, 0, stream>>>(attnL, stats, Vtb, Obf);
  // 7) output projection
  k_out_gemm<<<pg, 256, 0, stream>>>(Obf, wot, bo, out0);
}
